// WeightedSumGraphRepresentation_14448269984587
// MI455X (gfx1250) — matrix operand order repaired and checked by reading, not compiled, not run on hardware
//
#include <hip/hip_runtime.h>
#include <math.h>

#define FIN     256
#define GREP    256
#define HEADS   8
#define NGRAPH  1024
#define MTILE   128
#define PWW     32768   /* packed dwords per 256x256 weight */

typedef __attribute__((ext_vector_type(16))) __bf16 v16bf;
typedef __attribute__((ext_vector_type(8)))  float  v8f;

union ABFrag { unsigned u[8]; v16bf v; };

__device__ __forceinline__ unsigned short f2bf(float f) {
  unsigned u = __float_as_uint(f);
  unsigned r = (u + 0x7FFFu + ((u >> 16) & 1u)) >> 16;   // RNE
  return (unsigned short)r;
}
__device__ __forceinline__ float bf2f(unsigned short h) {
  return __uint_as_float(((unsigned)h) << 16);
}
// mish(x) = x * tanh(softplus(x)) = x * (e^2+2e)/(e^2+2e+2), e = exp(x)
// (for x > 20 the ratio is 1.0 in fp32; guard before e^2 can overflow at x ~ 44)
__device__ __forceinline__ float mishf(float x) {
  float e = __expf(x);
  float n = __fmaf_rn(e, e, 2.f * e);
  float r = n * __builtin_amdgcn_rcpf(n + 2.f);
  return (x > 20.f) ? x : x * r;
}
// monotone encoding of float for unsigned atomicMax
__device__ __forceinline__ unsigned encf(float f) {
  unsigned u = __float_as_uint(f);
  return (u & 0x80000000u) ? ~u : (u | 0x80000000u);
}
__device__ __forceinline__ float decf(unsigned e) {
  unsigned u = (e & 0x80000000u) ? (e ^ 0x80000000u) : ~e;
  return __uint_as_float(u);
}

// ---- one 16x16 output tile, K=256 as 8 x (16x16x32) bf16 WMMA steps ----
// A: wave's 16 rows from LDS (bf16 row-major, 256 cols). B: repacked global.
__device__ __forceinline__ v8f gemm_tile(const unsigned* ldsIn,
                                         const unsigned* packedW,
                                         float biasv, int lane, int rowbase, int j)
{
  v8f acc;
  #pragma unroll
  for (int i = 0; i < 8; ++i) acc[i] = biasv;
  const int half = lane >> 4;
  const int M    = rowbase + (lane & 15);
  #pragma unroll
  for (int kk = 0; kk < 8; ++kk) {
    ABFrag a, b;
    // A frag: VGPR i holds {act[M][k0], act[M][k0+1]}, k0 = 32kk + (i<4?0:16) + 8*half + 2*(i&3)
    // -> dword row index kk*16 + (i<4?0:8) + 4*half + (i&3): two contiguous b128 reads
    const uint4* ap = (const uint4*)(ldsIn + M * 128 + kk * 16 + 4 * half);
    uint4 a0 = ap[0];
    uint4 a1 = ap[2];
    a.u[0] = a0.x; a.u[1] = a0.y; a.u[2] = a0.z; a.u[3] = a0.w;
    a.u[4] = a1.x; a.u[5] = a1.y; a.u[6] = a1.z; a.u[7] = a1.w;
    // B frag: repacked fragment-major, 8 contiguous dwords per lane
    const uint4* bp = (const uint4*)(packedW + (size_t)(j * 8 + kk) * 256 + lane * 8);
    uint4 b0 = bp[0];
    uint4 b1 = bp[1];
    b.u[0] = b0.x; b.u[1] = b0.y; b.u[2] = b0.z; b.u[3] = b0.w;
    b.u[4] = b1.x; b.u[5] = b1.y; b.u[6] = b1.z; b.u[7] = b1.w;
    acc = __builtin_amdgcn_wmma_f32_16x16x32_bf16(false, a.v, false, b.v,
                                                  (short)0, acc, false, false);
  }
  return acc;
}

// one full 256->256 layer (bias + Mish), wave handles its own 16-row slab
__device__ __forceinline__ void mlp_layer256(const unsigned short* inAct,
                                             unsigned short* outAct,
                                             const unsigned* packedW, const float* bias,
                                             int lane, int wave)
{
  const unsigned* in32 = (const unsigned*)inAct;
  const int rowbase = wave * 16;
  const int l16 = lane & 15, half = lane >> 4;
  #pragma unroll 1
  for (int j = 0; j < 16; ++j) {
    float biasv = bias[16 * j + l16];
    v8f acc = gemm_tile(in32, packedW, biasv, lane, rowbase, j);
    #pragma unroll
    for (int v = 0; v < 8; ++v) {
      int r = rowbase + v + 8 * half;                 // C/D layout: VGPR v, half-lane
      outAct[r * 256 + 16 * j + l16] = f2bf(mishf(acc[v]));
    }
  }
}

__device__ __forceinline__ void load_x_tile(const float* X, unsigned short* act,
                                            int tilebase, int Nn, int tid)
{
  for (int i = tid; i < MTILE * 64; i += 256) {       // one float4 each
    int r = i >> 6;
    int c = (i & 63) << 2;
    int gr = tilebase + r;
    float4 f;
    if (gr < Nn) f = ((const float4*)(X + (size_t)gr * FIN))[c >> 2];
    else         f = make_float4(0.f, 0.f, 0.f, 0.f);
    unsigned short* o = act + r * 256 + c;
    o[0] = f2bf(f.x); o[1] = f2bf(f.y); o[2] = f2bf(f.z); o[3] = f2bf(f.w);
  }
}

// ---- repack one f32 weight matrix into bf16 WMMA-B fragment order ----
__global__ __launch_bounds__(256) void repack_kernel(const float* __restrict__ W,
                                                     unsigned* __restrict__ dst,
                                                     int ncols, int njt)
{
  int id = blockIdx.x * 256 + threadIdx.x;
  int total = njt * 2048;                 // njt*8*32*8
  if (id >= total) return;
  int d  = id & 7;
  int L  = (id >> 3) & 31;
  int kk = (id >> 8) & 7;
  int j  = id >> 11;
  int c  = 16 * j + (L & 15);
  int k0 = 32 * kk + (d < 4 ? 0 : 16) + 8 * (L >> 4) + 2 * (d & 3);
  float v0 = (c < ncols) ? W[(size_t)k0 * ncols + c]       : 0.f;
  float v1 = (c < ncols) ? W[(size_t)(k0 + 1) * ncols + c] : 0.f;
  dst[id] = (((unsigned)f2bf(v1)) << 16) | (unsigned)f2bf(v0);
}

__global__ __launch_bounds__(256) void init_kernel(float* out, unsigned* maxs, float* denom)
{
  int id = blockIdx.x * 256 + threadIdx.x;
  if (id < NGRAPH * GREP)   out[id]   = 0.f;
  if (id < HEADS)           maxs[id]  = 0u;          // encf of most-negative
  if (id < NGRAPH * HEADS)  denom[id] = 0.f;
}

// ---- fused 4-layer scoring MLP -> scores (N,8) ----
__global__ __launch_bounds__(256) void scores_kernel(
    const float* __restrict__ X,
    const unsigned* pw0, const unsigned* pw1, const unsigned* pw2, const unsigned* pw3,
    const float* b0, const float* b1, const float* b2, const float* b3,
    float* __restrict__ scores, int Nn)
{
  __shared__ unsigned short act0[MTILE * 256];
  __shared__ unsigned short act1[MTILE * 256];
  const int tid = threadIdx.x, lane = tid & 31, wave = tid >> 5;
  const int tilebase = blockIdx.x * MTILE;

  load_x_tile(X, act0, tilebase, Nn, tid);
  __syncthreads();
  mlp_layer256(act0, act1, pw0, b0, lane, wave); __syncthreads();
  mlp_layer256(act1, act0, pw1, b1, lane, wave); __syncthreads();
  mlp_layer256(act0, act1, pw2, b2, lane, wave); __syncthreads();

  // layer 3: 256 -> 8 (single 16-wide tile, cols 8..15 padded with zero weights)
  const int l16 = lane & 15, half = lane >> 4, rowbase = wave * 16;
  float biasv = (l16 < 8) ? b3[l16] : 0.f;
  v8f acc = gemm_tile((const unsigned*)act1, pw3, biasv, lane, rowbase, 0);
  if (l16 < 8) {
    #pragma unroll
    for (int v = 0; v < 8; ++v) {
      int r = tilebase + rowbase + v + 8 * half;
      if (r < Nn) scores[(size_t)r * HEADS + l16] = mishf(acc[v]);
    }
  }
}

// ---- per-head global max ----
__global__ __launch_bounds__(256) void max_kernel(const float* __restrict__ scores,
                                                  unsigned* maxs, int Nn)
{
  __shared__ unsigned lmax[HEADS];
  if (threadIdx.x < HEADS) lmax[threadIdx.x] = 0u;
  __syncthreads();
  float m[HEADS];
  #pragma unroll
  for (int h = 0; h < HEADS; ++h) m[h] = -3.4e38f;
  for (int n = blockIdx.x * 256 + threadIdx.x; n < Nn; n += gridDim.x * 256) {
    #pragma unroll
    for (int h = 0; h < HEADS; ++h) m[h] = fmaxf(m[h], scores[(size_t)n * HEADS + h]);
  }
  #pragma unroll
  for (int h = 0; h < HEADS; ++h) atomicMax(&lmax[h], encf(m[h]));
  __syncthreads();
  if (threadIdx.x < HEADS) atomicMax(&maxs[threadIdx.x], lmax[threadIdx.x]);
}

// ---- shift scores by max (in place) + per-graph denominators ----
__global__ __launch_bounds__(256) void denom_kernel(float* __restrict__ scores,
                                                    const unsigned* __restrict__ maxs,
                                                    const int* __restrict__ seg,
                                                    float* __restrict__ denom, int Nn)
{
  int n = blockIdx.x * 256 + threadIdx.x;
  if (n >= Nn) return;
  int g = seg[n];
  if (g < 0) g = 0; if (g >= NGRAPH) g = NGRAPH - 1;
  #pragma unroll
  for (int h = 0; h < HEADS; ++h) {
    float s = scores[(size_t)n * HEADS + h] - decf(maxs[h]);
    scores[(size_t)n * HEADS + h] = s;
    atomicAdd(&denom[g * HEADS + h], s);
  }
}

// ---- fused 4-layer transform MLP + weighted segmented sum into d_out ----
__global__ __launch_bounds__(256) void output_kernel(
    const float* __restrict__ X, const int* __restrict__ seg,
    const unsigned* pw0, const unsigned* pw1, const unsigned* pw2, const unsigned* pw3,
    const float* b0, const float* b1, const float* b2, const float* b3,
    const float* __restrict__ scores, const float* __restrict__ denom,
    float* __restrict__ out, int Nn)
{
  __shared__ unsigned short act0[MTILE * 256];
  __shared__ unsigned short act1[MTILE * 256];
  __shared__ float wts[MTILE * HEADS];
  __shared__ int   segs[MTILE];
  const int tid = threadIdx.x, lane = tid & 31, wave = tid >> 5;
  const int tilebase = blockIdx.x * MTILE;

  load_x_tile(X, act0, tilebase, Nn, tid);
  if (tid < MTILE) {
    int gr = tilebase + tid;
    int g = 0;
    if (gr < Nn) { g = seg[gr]; if (g < 0) g = 0; if (g >= NGRAPH) g = NGRAPH - 1; }
    segs[tid] = g;
    #pragma unroll
    for (int h = 0; h < HEADS; ++h)
      wts[tid * HEADS + h] = (gr < Nn)
          ? scores[(size_t)gr * HEADS + h] / denom[g * HEADS + h] : 0.f;
  }
  __syncthreads();
  mlp_layer256(act0, act1, pw0, b0, lane, wave); __syncthreads();
  mlp_layer256(act1, act0, pw1, b1, lane, wave); __syncthreads();
  mlp_layer256(act0, act1, pw2, b2, lane, wave); __syncthreads();
  mlp_layer256(act1, act0, pw3, b3, lane, wave); __syncthreads();  // H (bf16) in act0

  // column-owner segmented reduction: thread t owns output column t
  const int c = tid;
  const int h = c >> 5;                        // GREP/HEADS = 32
  float sum = 0.f;
  int curg = segs[0];
  for (int n = 0; n < MTILE; ++n) {
    int g = segs[n];
    if (g != curg) {
      atomicAdd(&out[(size_t)curg * GREP + c], sum);
      sum = 0.f; curg = g;
    }
    sum += wts[n * HEADS + h] * bf2f(act0[n * 256 + c]);
  }
  atomicAdd(&out[(size_t)curg * GREP + c], sum);
}

extern "C" void kernel_launch(void* const* d_in, const int* in_sizes, int n_in,
                              void* d_out, int out_size, void* d_ws, size_t ws_size,
                              hipStream_t stream)
{
  (void)n_in; (void)out_size; (void)ws_size;
  const float* X   = (const float*)d_in[0];
  const int*   seg = (const int*)d_in[1];
  const float* sW[4] = {(const float*)d_in[2],  (const float*)d_in[3],
                        (const float*)d_in[4],  (const float*)d_in[5]};
  const float* sB[4] = {(const float*)d_in[6],  (const float*)d_in[7],
                        (const float*)d_in[8],  (const float*)d_in[9]};
  const float* tW[4] = {(const float*)d_in[10], (const float*)d_in[11],
                        (const float*)d_in[12], (const float*)d_in[13]};
  const float* tB[4] = {(const float*)d_in[14], (const float*)d_in[15],
                        (const float*)d_in[16], (const float*)d_in[17]};
  const int Nn = in_sizes[0] / FIN;

  // workspace layout (dwords)
  unsigned* ws32 = (unsigned*)d_ws;
  size_t off = 0;
  unsigned* PS0 = ws32 + off; off += PWW;
  unsigned* PS1 = ws32 + off; off += PWW;
  unsigned* PS2 = ws32 + off; off += PWW;
  unsigned* PT0 = ws32 + off; off += PWW;
  unsigned* PT1 = ws32 + off; off += PWW;
  unsigned* PT2 = ws32 + off; off += PWW;
  unsigned* PT3 = ws32 + off; off += PWW;
  unsigned* PS3 = ws32 + off; off += 2048;
  float* scores = (float*)(ws32 + off); off += (size_t)Nn * HEADS;
  unsigned* maxs = ws32 + off;          off += HEADS;
  float* denom  = (float*)(ws32 + off);
  float* out = (float*)d_out;

  // 1) repack weights into WMMA-B fragment order (bf16 pairs)
  repack_kernel<<<128, 256, 0, stream>>>(sW[0], PS0, 256, 16);
  repack_kernel<<<128, 256, 0, stream>>>(sW[1], PS1, 256, 16);
  repack_kernel<<<128, 256, 0, stream>>>(sW[2], PS2, 256, 16);
  repack_kernel<<<  8, 256, 0, stream>>>(sW[3], PS3,   8,  1);
  repack_kernel<<<128, 256, 0, stream>>>(tW[0], PT0, 256, 16);
  repack_kernel<<<128, 256, 0, stream>>>(tW[1], PT1, 256, 16);
  repack_kernel<<<128, 256, 0, stream>>>(tW[2], PT2, 256, 16);
  repack_kernel<<<128, 256, 0, stream>>>(tW[3], PT3, 256, 16);

  // 2) zero accumulators
  init_kernel<<<(NGRAPH * GREP) / 256, 256, 0, stream>>>(out, maxs, denom);

  const int mtiles = (Nn + MTILE - 1) / MTILE;
  // 3) scoring MLP
  scores_kernel<<<mtiles, 256, 0, stream>>>(X, PS0, PS1, PS2, PS3,
                                            sB[0], sB[1], sB[2], sB[3], scores, Nn);
  // 4) per-head max
  max_kernel<<<256, 256, 0, stream>>>(scores, maxs, Nn);
  // 5) shift + per-graph denominators
  denom_kernel<<<(Nn + 255) / 256, 256, 0, stream>>>(scores, maxs, seg, denom, Nn);
  // 6) transform MLP + weighted segment sum
  output_kernel<<<mtiles, 256, 0, stream>>>(X, seg, PT0, PT1, PT2, PT3,
                                            tB[0], tB[1], tB[2], tB[3],
                                            scores, denom, out, Nn);
}
